// StandardGCN_26362509263608
// MI455X (gfx1250) — matrix operand order repaired and checked by reading, not compiled, not run on hardware
//
#include <hip/hip_runtime.h>
#include <hip/hip_bf16.h>

// ---------------------------------------------------------------------------
// StandardGCN pipeline for MI455X (gfx1250, wave32).
// Bandwidth-bound problem (~10.2 GFLOP GEMM vs ~150-250 MB traffic @ 23.3TB/s)
// -> fuse conv/BN/pool/adjacency/GCN into one pass, keep flat activations in
//    fp16, run the 1024x38912x128 classifier GEMM on v_wmma_f32_16x16x32_f16,
//    and K-slice the GEMM 4x (2048 waves, 304 WMMAs each) so the WMMA pipe has
//    enough independent waves to hide L2 latency on a device of MI455X's scale.
// ---------------------------------------------------------------------------

typedef _Float16 half8   __attribute__((ext_vector_type(8)));
typedef _Float16 half16  __attribute__((ext_vector_type(16)));
typedef float    floatx8 __attribute__((ext_vector_type(8)));

#define N_NODES  19
#define N_FILT   8
#define CONV_CH  152          // 19*8
#define T_IN     512
#define T_POOL   128
#define GCN_OUT  16
#define FLAT     38912        // 19*128*16
#define HID      128
#define BATCH    1024
#define BN_EPS   1e-5f
#define KSLICE   4
#define KCHUNK   (FLAT / KSLICE)   // 9728 = 304 * 32

// ---------------------------------------------------------------------------
// Kernel 1: row softmax of the 19x19 adjacency (tiny, one wave).
// ---------------------------------------------------------------------------
__global__ void k_softmax_adj(const float* __restrict__ adj, float* __restrict__ A)
{
    int r = threadIdx.x;
    if (r < N_NODES) {
        float m = -1e30f;
        for (int j = 0; j < N_NODES; ++j) m = fmaxf(m, adj[r * N_NODES + j]);
        float e[N_NODES];
        float s = 0.f;
        for (int j = 0; j < N_NODES; ++j) { e[j] = __expf(adj[r * N_NODES + j] - m); s += e[j]; }
        float inv = 1.f / s;
        for (int j = 0; j < N_NODES; ++j) A[r * N_NODES + j] = e[j] * inv;
    }
}

// ---------------------------------------------------------------------------
// Kernel 2: convert w1 (128 x 38912 fp32) to fp16 once; it then lives in L2.
// ---------------------------------------------------------------------------
__global__ void k_w1_to_f16(const float* __restrict__ w1, _Float16* __restrict__ w1h, int n)
{
    int i = blockIdx.x * blockDim.x + threadIdx.x;
    int stride = gridDim.x * blockDim.x;
    for (; i < n; i += stride) w1h[i] = (_Float16)w1[i];
}

// ---------------------------------------------------------------------------
// Kernel 3: per-batch fused conv(k=5, depthwise x8) + folded-BN + ReLU +
// MaxPool4 -> LDS [152x128] fp16, then adjacency mix (19-way) + GCN linear
// (8->16) + ReLU -> fp16 flat activations in reference flatten order.
// One block per batch element; static LDS ~45 KB.
// ---------------------------------------------------------------------------
__global__ __launch_bounds__(256) void k_features(
    const float* __restrict__ x,        // [B,19,512]
    const float* __restrict__ conv_w,   // [152,1,5]
    const float* __restrict__ conv_b,   // [152]
    const float* __restrict__ bn_gamma,
    const float* __restrict__ bn_beta,
    const float* __restrict__ bn_mean,
    const float* __restrict__ bn_var,
    const float* __restrict__ A,        // [19,19] softmaxed
    const float* __restrict__ gcn_w,    // [16,8]
    const float* __restrict__ gcn_b,    // [16]
    _Float16* __restrict__ flatH)       // [B,38912] fp16
{
    __shared__ _Float16 hp[CONV_CH * T_POOL];        // pooled features [c][t]
    __shared__ float sA[N_NODES * N_NODES];
    __shared__ float scw[CONV_CH * 5];
    __shared__ float ssc[CONV_CH];                   // BN scale
    __shared__ float ssh[CONV_CH];                   // BN shift (conv bias folded in)
    __shared__ float sgw[GCN_OUT * N_FILT];
    __shared__ float sgb[GCN_OUT];

    const int tid = threadIdx.x;
    const int b   = blockIdx.x;

    for (int i = tid; i < N_NODES * N_NODES; i += 256) sA[i] = A[i];
    for (int i = tid; i < CONV_CH * 5;       i += 256) scw[i] = conv_w[i];
    for (int i = tid; i < CONV_CH;           i += 256) {
        float sc = bn_gamma[i] * rsqrtf(bn_var[i] + BN_EPS);
        ssc[i] = sc;
        ssh[i] = bn_beta[i] + (conv_b[i] - bn_mean[i]) * sc;
    }
    for (int i = tid; i < GCN_OUT * N_FILT;  i += 256) sgw[i] = gcn_w[i];
    for (int i = tid; i < GCN_OUT;           i += 256) sgb[i] = gcn_b[i];
    __syncthreads();

    const float* xb = x + (size_t)b * N_NODES * T_IN;

    // ---- conv + affine + maxpool(4) + relu ----
    for (int p = tid; p < CONV_CH * T_POOL; p += 256) {
        const int c    = p / T_POOL;
        const int tp   = p % T_POOL;
        const int node = c / N_FILT;                 // depthwise group
        const float* xr = xb + node * T_IN;
        const float sc = ssc[c], sh = ssh[c];
        const int t0 = tp * 4;
        float m = -1e30f;
        #pragma unroll
        for (int q = 0; q < 4; ++q) {
            const int t = t0 + q;
            float s = 0.f;
            #pragma unroll
            for (int k = 0; k < 5; ++k) {
                const int ti = t + k - 2;            // padding (2,2)
                const float xv = (ti >= 0 && ti < T_IN) ? xr[ti] : 0.f;
                s += xv * scw[c * 5 + k];
            }
            m = fmaxf(m, s * sc + sh);               // affine before max
        }
        hp[c * T_POOL + tp] = (_Float16)fmaxf(m, 0.f);   // relu(max) == max(relu)
    }
    __syncthreads();

    // ---- adjacency aggregation + GCN linear + relu -> fp16 flat ----
    _Float16* fb = flatH + (size_t)b * FLAT;
    for (int pair = tid; pair < N_NODES * T_POOL; pair += 256) {
        const int n = pair / T_POOL;
        const int t = pair % T_POOL;
        float supp[N_FILT];
        #pragma unroll
        for (int f = 0; f < N_FILT; ++f) supp[f] = 0.f;
        for (int j = 0; j < N_NODES; ++j) {
            const float a = sA[n * N_NODES + j];
            #pragma unroll
            for (int f = 0; f < N_FILT; ++f)
                supp[f] += a * (float)hp[(j * N_FILT + f) * T_POOL + t];
        }
        half16 outv;
        #pragma unroll
        for (int o = 0; o < GCN_OUT; ++o) {
            float acc = sgb[o];
            #pragma unroll
            for (int f = 0; f < N_FILT; ++f) acc += supp[f] * sgw[o * N_FILT + f];
            outv[o] = (_Float16)fmaxf(acc, 0.f);
        }
        // flatten order: n*(128*16) + t*16 + o  -> 32B aligned vector store
        *(half16*)(fb + (size_t)n * (T_POOL * GCN_OUT) + t * GCN_OUT) = outv;
    }
}

// ---------------------------------------------------------------------------
// Kernel 4: partial[s] = flat[:, Ks] @ w1[:, Ks]^T  via WMMA f32_16x16x32_f16.
// M=1024, N=128, K=38912 split into KSLICE=4 slices of 9728 (304 WMMAs each).
// Grid: 512 blocks x 4 waves = 2048 waves. Slice = blockIdx/128 (uniform per
// block -> scalar loop bounds, no exec-mask loop); the wave id only picks the
// 16x16 output tile. K-offset folded into base pointers; trip count is a
// compile-time constant so the backend emits a scalar-counter loop.
// A lane layout (16-bit A 16x32): lanes 0-15 hold K 0-7 & 16-23 for row M=lane,
// lanes 16-31 hold K 8-15 & 24-31. B (32x16, K-major per column): lane n%16 is
// column, lane/16 selects K 0-7 & 16-23 / 8-15 & 24-31 (same as A), from the w1 row.
// ---------------------------------------------------------------------------
__global__ __launch_bounds__(128) void k_gemm_wmma(
    const _Float16* __restrict__ Af,      // [1024, 38912] activations
    const _Float16* __restrict__ Bw,      // [128, 38912]  w1 fp16 (row = output ch)
    float*          __restrict__ partial) // [KSLICE, 1024, 128] fp32
{
    const int K     = FLAT;
    const int tid   = threadIdx.x;
    const int lane  = tid & 31;
    const int wave  = tid >> 5;
    const int slice = blockIdx.x >> 7;              // 0..3   (uniform per block)
    const int tile  = ((blockIdx.x & 127) << 2) + wave;  // 0..511
    const int mt    = tile >> 3;                    // 64 M-tiles
    const int nt    = tile & 7;                     // 8  N-tiles
    const int kbeg  = slice * KCHUNK;               // scalar

    const int arow = lane & 15;
    const int akb  = (lane >> 4) * 8;               // K sub-block base for A
    const _Float16* aptr = Af + (size_t)(mt * 16 + arow) * K + kbeg + akb;

    const int bcol = lane & 15;
    const int bkg  = (lane >> 4) * 8;               // K sub-block base for B
    const _Float16* bptr = Bw + (size_t)(nt * 16 + bcol) * K + kbeg + bkg;

    floatx8 c = {};
    #pragma unroll 4
    for (int kk = 0; kk < KCHUNK; kk += 32) {       // constant trip count: 304
        half8 alo = *(const half8*)(aptr + kk);        // K = base+0..7
        half8 ahi = *(const half8*)(aptr + kk + 16);   // K = base+16..23
        half16 a  = __builtin_shufflevector(alo, ahi,
                     0,1,2,3,4,5,6,7,8,9,10,11,12,13,14,15);
        half8 blo = *(const half8*)(bptr + kk);        // K = base+0..7
        half8 bhi = *(const half8*)(bptr + kk + 16);   // K = base+16..23
        half16 bm = __builtin_shufflevector(blo, bhi,
                     0,1,2,3,4,5,6,7,8,9,10,11,12,13,14,15);
        c = __builtin_amdgcn_wmma_f32_16x16x32_f16(
                /*neg_a=*/false, a, /*neg_b=*/false, bm,
                /*c_mod=*/(short)0, c, /*reuse_a=*/false, /*reuse_b=*/false);
    }

    // C/D layout: VGPR r -> M = r + (lane/16)*8, N = lane%16
    const int nglob = nt * 16 + (lane & 15);
    const int mbase = mt * 16 + (lane >> 4) * 8;
    float* pdst = partial + (size_t)slice * BATCH * HID;
    #pragma unroll
    for (int r = 0; r < 8; ++r)
        pdst[(size_t)(mbase + r) * HID + nglob] = c[r];
}

// ---------------------------------------------------------------------------
// Kernel 5: reduce K-slices, apply bias+ReLU, then 128->2 classifier.
// out[b,:] = relu(sum_s partial[s,b,:] + b1) @ w2^T + b2
// ---------------------------------------------------------------------------
__global__ void k_final(const float* __restrict__ partial,
                        const float* __restrict__ b1,
                        const float* __restrict__ w2,
                        const float* __restrict__ b2,
                        float* __restrict__ out)
{
    int b = blockIdx.x * blockDim.x + threadIdx.x;
    if (b >= BATCH) return;
    const float* p = partial + (size_t)b * HID;
    float a0 = b2[0], a1 = b2[1];
    #pragma unroll 4
    for (int m = 0; m < HID; ++m) {
        float h = p[m]
                + p[m + 1 * BATCH * HID]
                + p[m + 2 * BATCH * HID]
                + p[m + 3 * BATCH * HID]
                + b1[m];
        h = fmaxf(h, 0.f);
        a0 += h * w2[m];
        a1 += h * w2[HID + m];
    }
    out[2 * b + 0] = a0;
    out[2 * b + 1] = a1;
}

// ---------------------------------------------------------------------------
// Launch
// ---------------------------------------------------------------------------
extern "C" void kernel_launch(void* const* d_in, const int* in_sizes, int n_in,
                              void* d_out, int out_size, void* d_ws, size_t ws_size,
                              hipStream_t stream)
{
    const float* x    = (const float*)d_in[0];
    const float* cw   = (const float*)d_in[1];
    const float* cb   = (const float*)d_in[2];
    const float* bg   = (const float*)d_in[3];
    const float* bb   = (const float*)d_in[4];
    const float* bm   = (const float*)d_in[5];
    const float* bv   = (const float*)d_in[6];
    const float* adj  = (const float*)d_in[7];
    const float* gw   = (const float*)d_in[8];
    const float* gb   = (const float*)d_in[9];
    const float* w1   = (const float*)d_in[10];
    const float* b1   = (const float*)d_in[11];
    const float* w2   = (const float*)d_in[12];
    const float* b2   = (const float*)d_in[13];
    float* out = (float*)d_out;

    // Workspace layout (all offsets 256B-aligned):
    //   [0,      4096)            softmax(adj) fp32
    //   [4096,   +2MB)            partial      fp32 [4,1024,128]
    //   [+0,     +9.96MB)         w1 fp16      [128,38912]
    //   [+0,     +79.7MB)         flat fp16    [1024,38912]
    char* ws = (char*)d_ws;
    float*    A       = (float*)ws;
    float*    partial = (float*)(ws + 4096);
    _Float16* w1h     = (_Float16*)(ws + 4096 + (size_t)KSLICE * BATCH * HID * 4);
    _Float16* flatH   = (_Float16*)(ws + 4096 + (size_t)KSLICE * BATCH * HID * 4
                                            + (size_t)HID * FLAT * 2);

    k_softmax_adj<<<1, 32, 0, stream>>>(adj, A);
    k_w1_to_f16<<<2048, 256, 0, stream>>>(w1, w1h, HID * FLAT);
    k_features<<<BATCH, 256, 0, stream>>>(x, cw, cb, bg, bb, bm, bv, A, gw, gb, flatH);
    k_gemm_wmma<<<512, 128, 0, stream>>>(flatH, w1h, partial);
    k_final<<<(BATCH + 255) / 256, 256, 0, stream>>>(partial, b1, w2, b2, out);
}
